// ExportableGPTWithCache_14671608283563
// MI455X (gfx1250) — matrix operand order repaired and checked by reading, not compiled, not run on hardware
//
#include <hip/hip_runtime.h>
#include <hip/hip_bf16.h>
#include <math.h>

#define NL 12
#define T 1024
#define D 1024
#define H 8
#define HD 128
#define FF 4096
#define NV 32768
#define SC 4096
#define RMS_EPS 1.1920929e-07f

typedef __attribute__((ext_vector_type(16))) __bf16 v16bf;
typedef __attribute__((ext_vector_type(8)))  float  v8f;

// round-to-nearest f32->bf16; pack (a,b) -> {lo16=a, hi16=b} with one v_perm_b32
__device__ __forceinline__ unsigned pack_bf16(float a, float b) {
  unsigned ua = __float_as_uint(a) + 0x8000u;
  unsigned ub = __float_as_uint(b) + 0x8000u;
  return __builtin_amdgcn_perm(ub, ua, 0x07060302u);  // [ub.b3 ub.b2 ua.b3 ua.b2]
}
__device__ __forceinline__ unsigned short bf16_1(float a) {
  return (unsigned short)((__float_as_uint(a) + 0x8000u) >> 16);
}

__device__ __forceinline__ float blk_sum(float v, float* sred) {
  #pragma unroll
  for (int o = 16; o > 0; o >>= 1) v += __shfl_xor(v, o, 32);
  __syncthreads();
  if ((threadIdx.x & 31) == 0) sred[threadIdx.x >> 5] = v;
  __syncthreads();
  float s = sred[0];
  #pragma unroll
  for (int i = 1; i < 8; ++i) s += sred[i];
  return s;
}

__device__ __forceinline__ float blk_max(float v, float* sred) {
  #pragma unroll
  for (int o = 16; o > 0; o >>= 1) v = fmaxf(v, __shfl_xor(v, o, 32));
  __syncthreads();
  if ((threadIdx.x & 31) == 0) sred[threadIdx.x >> 5] = v;
  __syncthreads();
  float s = sred[0];
  #pragma unroll
  for (int i = 1; i < 8; ++i) s = fmaxf(s, sred[i]);
  return s;
}

// ---------------------------------------------------------------------------
// Generic batched WMMA GEMM: C = epi(A @ B (+R)), bf16 compute / fp32 I/O.
// BM=128, BN=128, BK=32; 8 waves, each wave owns a 64x32 output tile.
// Double-buffered LDS staging in a contiguous-k fragment order shared by A and B:
//  A 16x32 subtile: lane = m + 16*(k>=16);    elem = k & 15
//  B 32x16 subtile: lane = n + 16*(k>=16);    elem = k & 15
// causal: 0=none, 1=skip fully-masked blocks (scores), 2=clamp K to m0+BM (attn@V)
// ---------------------------------------------------------------------------
#define BM 128
#define BN 128
#define BK 32

__global__ __launch_bounds__(256)
void gemm_wmma(const float* A, int lda, long long batchA,
               const float* B, int ldb, long long batchB,
               float* C, int ldc, long long batchC,
               const float* R, int ldr, long long batchR,
               int K, int transB, int epi, int causal)
{
  __shared__ unsigned short As[2][BM * BK];  // 2 x 8 KB
  __shared__ unsigned short Bs[2][BK * BN];  // 2 x 8 KB

  const int tid  = threadIdx.x;
  const int lane = tid & 31;
  const int wv   = tid >> 5;
  const int m0   = blockIdx.y * BM;
  const int n0   = blockIdx.x * BN;

  if (causal == 1 && n0 > m0 + BM - 1) return;     // fully masked score block
  int Keff = K;
  if (causal == 2) { int lim = m0 + BM; Keff = (lim < K) ? lim : K; }

  const long long z = blockIdx.z;
  A += z * batchA;
  B += z * batchB;
  C += z * batchC;
  if (R) R += z * batchR;

  const int mW = (wv & 1) * 64;   // 2 wave-rows
  const int nW = (wv >> 1) * 32;  // 4 wave-cols

  // per-thread staging registers (A: float4 along K; B: float4 per mode)
  float4 aR[4];
  float4 bR[4];

  auto load_tile = [&](int k0) {
    #pragma unroll
    for (int p = 0; p < 4; ++p) {
      int i4  = tid + p * 256;      // 0..1023 float4s
      int row = i4 >> 3;            // 0..127
      int kk  = (i4 & 7) * 4;       // 0..28
      aR[p] = *(const float4*)(A + (long long)(m0 + row) * lda + (k0 + kk));
    }
    if (!transB) {
      #pragma unroll
      for (int p = 0; p < 4; ++p) {
        int i4 = tid + p * 256;
        int kk = i4 >> 5;            // 0..31
        int n  = (i4 & 31) * 4;      // 0..124
        bR[p] = *(const float4*)(B + (long long)(k0 + kk) * ldb + (n0 + n));
      }
    } else {
      #pragma unroll
      for (int p = 0; p < 4; ++p) {
        int i4 = tid + p * 256;
        int n  = i4 >> 3;            // 0..127
        int kk = (i4 & 7) * 4;       // 0..28
        bR[p] = *(const float4*)(B + (long long)(n0 + n) * ldb + (k0 + kk));
      }
    }
  };

  auto store_tile = [&](unsigned short* as, unsigned short* bs) {
    // A: 4 consecutive K per thread -> 2 packed dwords, one 8B-aligned b64 store
    #pragma unroll
    for (int p = 0; p < 4; ++p) {
      int i4  = tid + p * 256;
      int row = i4 >> 3;
      int kk  = (i4 & 7) * 4;
      int smt   = row >> 4;
      int m     = row & 15;
      int khalf = (kk >> 4) & 1;
      int vv0   = (kk & 15) >> 1;
      int addr  = smt * 512 + (m + (khalf << 4)) * 16 + vv0 * 2;  // ushort idx, 8B aligned
      uint2 u;
      u.x = pack_bf16(aR[p].x, aR[p].y);
      u.y = pack_bf16(aR[p].z, aR[p].w);
      *(uint2*)&as[addr] = u;
    }
    if (!transB) {
      // values run along N -> scatter to 4 lanes with b16 stores
      #pragma unroll
      for (int p = 0; p < 4; ++p) {
        int i4 = tid + p * 256;
        int kk = i4 >> 5;
        int n  = (i4 & 31) * 4;
        float vals[4] = {bR[p].x, bR[p].y, bR[p].z, bR[p].w};
        int khalf = (kk >= 16) ? 16 : 0;
        int e     = kk & 15;
        #pragma unroll
        for (int j = 0; j < 4; ++j) {
          int nn = n + j;
          bs[(nn >> 4) * 512 + ((nn & 15) + khalf) * 16 + e] = bf16_1(vals[j]);
        }
      }
    } else {
      // values run along K -> 2 packed dwords, one b64 store
      #pragma unroll
      for (int p = 0; p < 4; ++p) {
        int i4 = tid + p * 256;
        int n  = i4 >> 3;
        int kk = (i4 & 7) * 4;
        int khalf = (kk >= 16) ? 16 : 0;
        int addr  = (n >> 4) * 512 + ((n & 15) + khalf) * 16 + (kk & 15);
        uint2 u;
        u.x = pack_bf16(bR[p].x, bR[p].y);
        u.y = pack_bf16(bR[p].z, bR[p].w);
        *(uint2*)&bs[addr] = u;
      }
    }
  };

  v8f acc[4][2];
  #pragma unroll
  for (int i = 0; i < 4; ++i)
    #pragma unroll
    for (int j = 0; j < 2; ++j)
      acc[i][j] = (v8f){0.f,0.f,0.f,0.f,0.f,0.f,0.f,0.f};

  const int nk = Keff / BK;
  // prologue: stage tile 0 into buffer 0
  load_tile(0);
  store_tile(As[0], Bs[0]);

  for (int t = 0; t < nk; ++t) {
    __syncthreads();                       // staging of buf[cur] complete
    const int cur = t & 1;
    const bool more = (t + 1 < nk);
    if (more) load_tile((t + 1) * BK);     // prefetch next tile (VMEM in flight)

    // fragment loads (2x ds_load_b128 each) + 8 WMMAs
    v16bf af[4], bfr[2];
    #pragma unroll
    for (int i = 0; i < 4; ++i)
      af[i] = *(const v16bf*)&As[cur][((mW >> 4) + i) * 512 + lane * 16];
    #pragma unroll
    for (int j = 0; j < 2; ++j)
      bfr[j] = *(const v16bf*)&Bs[cur][((nW >> 4) + j) * 512 + lane * 16];
    #pragma unroll
    for (int i = 0; i < 4; ++i)
      #pragma unroll
      for (int j = 0; j < 2; ++j)
        acc[i][j] = __builtin_amdgcn_wmma_f32_16x16x32_bf16(
            false, af[i], false, bfr[j], (short)0, acc[i][j], false, false);

    if (more) store_tile(As[1 - cur], Bs[1 - cur]);  // stage next into other buffer
  }

  // ---- epilogue: C layout lane=(n), VGPR v -> row v + 8*(lane>>4)
  #pragma unroll
  for (int i = 0; i < 4; ++i) {
    #pragma unroll
    for (int j = 0; j < 2; ++j) {
      int colg = n0 + nW + j * 16 + (lane & 15);
      int rowb = m0 + mW + i * 16 + ((lane >> 4) << 3);
      #pragma unroll
      for (int vv = 0; vv < 8; ++vv) {
        int rowg = rowb + vv;
        float val = acc[i][j][vv];
        if (epi == 1)      val += R[(long long)rowg * ldr + colg];
        else if (epi == 2) { float r = fmaxf(val, 0.f); val = r * r; }
        else if (epi == 3) val = 15.0f * tanhf(val * (1.0f / 15.0f));
        C[(long long)rowg * ldc + colg] = val;
      }
    }
  }
}

// ---------------------------------------------------------------------------
__global__ __launch_bounds__(256)
void embed_rms_kernel(const int* ids, const float* wte, float* x) {
  __shared__ float sred[8];
  int t = blockIdx.x;
  const float* row = wte + (long long)ids[t] * D;
  float vals[4]; float ss = 0.f;
  #pragma unroll
  for (int i = 0; i < 4; ++i) { vals[i] = row[threadIdx.x + i * 256]; ss += vals[i] * vals[i]; }
  ss = blk_sum(ss, sred);
  float r = rsqrtf(ss * (1.0f / D) + RMS_EPS);
  #pragma unroll
  for (int i = 0; i < 4; ++i) x[(long long)t * D + threadIdx.x + i * 256] = vals[i] * r;
}

__global__ __launch_bounds__(256)
void rms_rows_kernel(const float* in, float* out) {
  __shared__ float sred[8];
  long long base = (long long)blockIdx.x * D;
  float vals[4]; float ss = 0.f;
  #pragma unroll
  for (int i = 0; i < 4; ++i) { vals[i] = in[base + threadIdx.x + i * 256]; ss += vals[i] * vals[i]; }
  ss = blk_sum(ss, sred);
  float r = rsqrtf(ss * (1.0f / D) + RMS_EPS);
  #pragma unroll
  for (int i = 0; i < 4; ++i) out[base + threadIdx.x + i * 256] = vals[i] * r;
}

// one wave per (token, head): RoPE then per-head RMS, in-place
__global__ __launch_bounds__(256)
void rope_rms_kernel(float* q, const float* cosb, const float* sinb) {
  int t = blockIdx.x;
  int h = threadIdx.x >> 5;
  int lane = threadIdx.x & 31;
  float* p = q + (long long)t * (H * HD) + h * HD;
  const float* cr = cosb + (long long)t * 64;
  const float* sr = sinb + (long long)t * 64;
  float o[4]; float ss = 0.f;
  #pragma unroll
  for (int u = 0; u < 2; ++u) {
    int i = lane + u * 32;
    float c = cr[i], s = sr[i];
    float x1 = p[i], x2 = p[i + 64];
    float a = x1 * c + x2 * s;
    float b = -x1 * s + x2 * c;
    o[u * 2] = a; o[u * 2 + 1] = b;
    ss += a * a + b * b;
  }
  #pragma unroll
  for (int off = 16; off > 0; off >>= 1) ss += __shfl_xor(ss, off, 32);
  float r = rsqrtf(ss * (1.0f / HD) + RMS_EPS);
  #pragma unroll
  for (int u = 0; u < 2; ++u) {
    int i = lane + u * 32;
    p[i] = o[u * 2] * r;
    p[i + 64] = o[u * 2 + 1] * r;
  }
}

__global__ __launch_bounds__(256)
void softmax_causal_kernel(float* scores) {
  __shared__ float sred[8];
  int row = blockIdx.x, h = blockIdx.y;
  float* p = scores + ((long long)h * T + row) * T;
  const float scale = 0.08838834764831845f; // 1/sqrt(128)
  float v[4];
  float mx = -__builtin_inff();
  #pragma unroll
  for (int i = 0; i < 4; ++i) {
    int c = threadIdx.x + i * 256;
    v[i] = (c <= row) ? p[c] * scale : -__builtin_inff();
    mx = fmaxf(mx, v[i]);
  }
  mx = blk_max(mx, sred);
  float ss = 0.f;
  #pragma unroll
  for (int i = 0; i < 4; ++i) {
    int c = threadIdx.x + i * 256;
    v[i] = (c <= row) ? __expf(v[i] - mx) : 0.f;
    ss += v[i];
  }
  ss = blk_sum(ss, sred);
  float inv = 1.0f / ss;
  #pragma unroll
  for (int i = 0; i < 4; ++i) p[threadIdx.x + i * 256] = v[i] * inv;
}

__global__ void copy_f4_kernel(const float4* in, float4* out, long long n4) {
  long long i = (long long)blockIdx.x * 256 + threadIdx.x;
  if (i < n4) out[i] = in[i];
}

// src[t][h][d] (t stride 1024) -> dst_layer[h][pos+t][d] (h stride SC*HD)
__global__ __launch_bounds__(256)
void cache_scatter_kernel(const float* src, float* dst_layer, const int* pos) {
  int idx = blockIdx.x * 256 + threadIdx.x;  // 0 .. T*H*HD-1
  int d = idx & (HD - 1);
  int h = (idx >> 7) & (H - 1);
  int t = idx >> 10;
  int tp = pos[0] + t;
  if (tp < SC)
    dst_layer[((long long)h * SC + tp) * HD + d] = src[idx];
}

// ---------------------------------------------------------------------------
static void launch_gemm(hipStream_t st,
                        const float* A, int lda, long long bA,
                        const float* B, int ldb, long long bB,
                        float* C, int ldc, long long bC,
                        const float* R, int ldr, long long bR,
                        int M, int N, int K, int transB, int epi, int batch,
                        int causal = 0) {
  dim3 g(N / BN, M / BM, batch), b(256);
  gemm_wmma<<<g, b, 0, st>>>(A, lda, bA, B, ldb, bB, C, ldc, bC, R, ldr, bR,
                             K, transB, epi, causal);
}

extern "C" void kernel_launch(void* const* d_in, const int* in_sizes, int n_in,
                              void* d_out, int out_size, void* d_ws, size_t ws_size,
                              hipStream_t stream) {
  (void)in_sizes; (void)n_in; (void)out_size; (void)ws_size;
  const int*   ids   = (const int*)d_in[0];
  const float* wte   = (const float*)d_in[1];
  const float* wq    = (const float*)d_in[2];
  const float* wk    = (const float*)d_in[3];
  const float* wvw   = (const float*)d_in[4];
  const float* wo    = (const float*)d_in[5];
  const float* wfc   = (const float*)d_in[6];
  const float* wproj = (const float*)d_in[7];
  const float* lmw   = (const float*)d_in[8];
  const float* cosb  = (const float*)d_in[9];
  const float* sinb  = (const float*)d_in[10];
  const float* ck_in = (const float*)d_in[11];
  const float* cv_in = (const float*)d_in[12];
  const int*   pos   = (const int*)d_in[13];

  float* out    = (float*)d_out;
  float* logits = out;                           // T*NV = 33,554,432
  float* ck_out = out + 33554432LL;              // NL*H*SC*HD = 50,331,648
  float* cv_out = ck_out + 50331648LL;

  float* w  = (float*)d_ws;                      // needs 18M floats (72 MB)
  float* x  = w;                                 // T*D
  float* h  = w + 1 * 1048576;                   // T*D
  float* q  = w + 2 * 1048576;                   // T*H*HD
  float* k  = w + 3 * 1048576;
  float* v  = w + 4 * 1048576;
  float* y  = w + 5 * 1048576;                   // T*D
  float* ff = w + 6 * 1048576;                   // T*FF (4M)
  float* sc = w + 10 * 1048576;                  // H*T*T (8M)

  embed_rms_kernel<<<T, 256, 0, stream>>>(ids, wte, x);
  {
    long long n4 = 50331648LL / 4;
    int nb = (int)((n4 + 255) / 256);
    copy_f4_kernel<<<nb, 256, 0, stream>>>((const float4*)ck_in, (float4*)ck_out, n4);
    copy_f4_kernel<<<nb, 256, 0, stream>>>((const float4*)cv_in, (float4*)cv_out, n4);
  }

  for (int l = 0; l < NL; ++l) {
    const float* wq_l = wq    + (long long)l * 1048576;
    const float* wk_l = wk    + (long long)l * 1048576;
    const float* wv_l = wvw   + (long long)l * 1048576;
    const float* wo_l = wo    + (long long)l * 1048576;
    const float* wf_l = wfc   + (long long)l * 4194304;
    const float* wp_l = wproj + (long long)l * 4194304;

    rms_rows_kernel<<<T, 256, 0, stream>>>(x, h);
    launch_gemm(stream, h, D, 0, wq_l, D, 0, q, D, 0, nullptr, 0, 0, T, D, D, 0, 0, 1);
    launch_gemm(stream, h, D, 0, wk_l, D, 0, k, D, 0, nullptr, 0, 0, T, D, D, 0, 0, 1);
    launch_gemm(stream, h, D, 0, wv_l, D, 0, v, D, 0, nullptr, 0, 0, T, D, D, 0, 0, 1);
    rope_rms_kernel<<<T, 256, 0, stream>>>(q, cosb, sinb);
    rope_rms_kernel<<<T, 256, 0, stream>>>(k, cosb, sinb);
    cache_scatter_kernel<<<4096, 256, 0, stream>>>(k, ck_out + (long long)l * 4194304, pos);
    cache_scatter_kernel<<<4096, 256, 0, stream>>>(v, cv_out + (long long)l * 4194304, pos);

    // scores[h] = q_h (T x HD) @ k_h^T : batched over heads; skip masked blocks
    launch_gemm(stream, q, D, HD, k, D, HD, sc, T, (long long)T * T,
                nullptr, 0, 0, T, T, HD, 1, 0, H, 1);
    softmax_causal_kernel<<<dim3(T, H), 256, 0, stream>>>(sc);
    // y[:, h*HD:(h+1)*HD] = attn_h @ v_h ; K clamped to m0+BM (causal zeros)
    launch_gemm(stream, sc, T, (long long)T * T, v, D, HD, y, D, HD,
                nullptr, 0, 0, T, HD, T, 0, 0, H, 2);
    // x = x + y @ wo
    launch_gemm(stream, y, D, 0, wo_l, D, 0, x, D, 0, x, D, 0, T, D, D, 0, 1, 1);
    // MLP: ff = relu(rms(x) @ wfc)^2 ; x = x + ff @ wproj
    rms_rows_kernel<<<T, 256, 0, stream>>>(x, h);
    launch_gemm(stream, h, D, 0, wf_l, FF, 0, ff, FF, 0, nullptr, 0, 0, T, FF, D, 0, 2, 1);
    launch_gemm(stream, ff, FF, 0, wp_l, D, 0, x, D, 0, x, D, 0, T, D, FF, 0, 1, 1);
  }

  rms_rows_kernel<<<T, 256, 0, stream>>>(x, h);
  // logits = 15*tanh((rms(x) @ lm_w)/15)
  launch_gemm(stream, h, D, 0, lmw, NV, 0, logits, NV, 0, nullptr, 0, 0, T, NV, D, 0, 3, 1);
}
